// EncoderLayer_18622978195985
// MI455X (gfx1250) — matrix operand order repaired and checked by reading, not compiled, not run on hardware
//
#include <hip/hip_runtime.h>
#include <hip/hip_bf16.h>

// ---------------------------------------------------------------------------
// Transformer encoder layer for MI455X (gfx1250, wave32, WMMA).
// GEMMs: LDS-tiled 128x128 block / 32x64 per wave, double-buffered
//        global_load_async_to_lds_b128 staging (ASYNCcnt), WMMA f16->f32.
// Attention: flash-style online softmax; K/V tiles cooperatively staged in
//        double-buffered LDS via async DMA and shared by all 8 waves.
// ---------------------------------------------------------------------------

typedef __attribute__((ext_vector_type(16))) _Float16 half16;
typedef __attribute__((ext_vector_type(8)))  _Float16 half8;
typedef __attribute__((ext_vector_type(8)))  float    v8f;

constexpr int BATCH = 4;
constexpr int SEQ   = 2048;
constexpr int DM    = 512;     // d_model
constexpr int NH    = 8;       // heads
constexpr int DEPTH = 64;      // d_model / heads
constexpr int DFF   = 2048;
constexpr int ROWS  = BATCH * SEQ;   // 8192

static __device__ __forceinline__ half16 cat16(half8 lo, half8 hi) {
  return __builtin_shufflevector(lo, hi, 0,1,2,3,4,5,6,7,8,9,10,11,12,13,14,15);
}

// A-matrix fragment (16xK row-major source). ISA 7.12.2: lane (r = lane&15,
// h = lane>>4) holds K = [8h..8h+7] in elements 0..7 and K = [16+8h..23+8h]
// in elements 8..15, for row r.  -> two 16B loads.
static __device__ __forceinline__ half16
load_frag_a(const _Float16* __restrict__ base, int ld, int k0, int lane) {
  int r = lane & 15, h = lane >> 4;
  const _Float16* p = base + (size_t)r * ld + k0 + 8 * h;
  half8 lo = *(const half8*)p;
  half8 hi = *(const half8*)(p + 16);
  return cat16(lo, hi);
}

static __device__ __forceinline__ v8f wmma_f32_f16(half16 a, half16 b, v8f c) {
  // (neg_a, A, neg_b, B, c_mod, C, reuse_a, reuse_b)
  return __builtin_amdgcn_wmma_f32_16x16x32_f16(false, a, false, b,
                                                (short)0, c, false, false);
}

// ---------------------------------------------------------------------------
// CDNA5 async global->LDS copy (VGLOBAL opcode 98, tracked by ASYNCcnt).
// Low 32 bits of a generic LDS pointer are the DS byte address (ISA 10.2).
// ---------------------------------------------------------------------------
static __device__ __forceinline__ void async_b128(const void* gaddr,
                                                  uint32_t lds_addr) {
  asm volatile("global_load_async_to_lds_b128 %0, %1, off"
               :: "v"(lds_addr), "v"(gaddr) : "memory");
}
static __device__ __forceinline__ void wait_async_le4() {
  asm volatile("s_wait_asynccnt 0x4" ::: "memory");
}
static __device__ __forceinline__ void wait_async_le2() {
  asm volatile("s_wait_asynccnt 0x2" ::: "memory");
}
static __device__ __forceinline__ void wait_async_0() {
  asm volatile("s_wait_asynccnt 0x0" ::: "memory");
}

// ---------------------------------------------------------------------------
// Elementwise helpers
// ---------------------------------------------------------------------------
__global__ void cvt_f16_kernel(const float* __restrict__ src,
                               _Float16* __restrict__ dst, int n) {
  int i = blockIdx.x * blockDim.x + threadIdx.x;
  if (i < n) dst[i] = (_Float16)src[i];
}

// src: f32 [R][C] row-major -> dst: f16 [C][R] (i.e. transposed)
__global__ void transpose_cvt_kernel(const float* __restrict__ src,
                                     _Float16* __restrict__ dst, int R, int C) {
  int i = blockIdx.x * blockDim.x + threadIdx.x;
  if (i < R * C) {
    int r = i / C, c = i % C;
    dst[(size_t)c * R + r] = (_Float16)src[i];
  }
}

// vh: f16 [B,S,NH,DEPTH] -> vt: f16 [B,NH,DEPTH,S]
__global__ void transpose_v_kernel(const _Float16* __restrict__ vh,
                                   _Float16* __restrict__ vt) {
  int i = blockIdx.x * blockDim.x + threadIdx.x;   // over ROWS*DM
  if (i < ROWS * DM) {
    int d  = i & 63;
    int hd = (i >> 6) & 7;
    int s  = (i >> 9) & (SEQ - 1);
    int b  = i >> 20;
    vt[((size_t)((b * NH + hd) * DEPTH + d)) * SEQ + s] = vh[i];
  }
}

// ---------------------------------------------------------------------------
// WMMA GEMM: C[M][N] = A[M][K](f16,row-major) * Bt[N][K](f16)^T + bias.
// Block: 256 threads = 8 waves -> 128x128 tile (wave tile 32x64, 8 accums).
// K-loop in steps of 32 with double-buffered async global->LDS panels:
//   Ab: 128x32 row-major, Bb: 128 cols x 32 (column-major source slice).
// Per stage each thread issues 4 async b128 (32B of A + 32B of B).
// ---------------------------------------------------------------------------
__global__ __launch_bounds__(256)
void gemm_f16_kernel(const _Float16* __restrict__ A,
                     const _Float16* __restrict__ Bt,
                     const float*    __restrict__ bias,
                     float*          __restrict__ Cf,
                     _Float16*       __restrict__ Ch,
                     int M, int N, int K, int relu) {
  __shared__ __align__(16) _Float16 Ab[2][128 * 32];
  __shared__ __align__(16) _Float16 Bb[2][128 * 32];

  int tid  = threadIdx.x;
  int lane = tid & 31;
  int w    = tid >> 5;          // wave 0..7
  int wm   = w >> 1;            // 0..3  (row group of 32)
  int wn   = w & 1;             // 0..1  (col group of 64)

  int nbn = N >> 7;
  int bm  = blockIdx.x / nbn;
  int bn  = blockIdx.x % nbn;
  int tm0 = bm << 7;
  int bn0 = bn << 7;

  // global->LDS fill mapping: thread -> (panel row, two 16B segments)
  int frow = tid >> 1;               // 0..127
  int fseg = (tid & 1) * 2;          // segments {0,1} or {2,3} of a 64B row

  auto issue = [&](int buf, int k0) {
    const _Float16* ga = A  + (size_t)(tm0 + frow) * K + k0 + fseg * 8;
    const _Float16* gb = Bt + (size_t)(bn0 + frow) * K + k0 + fseg * 8;
    uint32_t la = (uint32_t)(size_t)&Ab[buf][frow * 32 + fseg * 8];
    uint32_t lb = (uint32_t)(size_t)&Bb[buf][frow * 32 + fseg * 8];
    async_b128(ga,     la);
    async_b128(ga + 8, la + 16);
    async_b128(gb,     lb);
    async_b128(gb + 8, lb + 16);
  };

  int ar = lane & 15, h = lane >> 4;

  v8f acc[2][4];
  v8f zero = {};
#pragma unroll
  for (int i = 0; i < 2; ++i)
#pragma unroll
    for (int j = 0; j < 4; ++j) acc[i][j] = zero;

  int nsteps = K >> 5;
  issue(0, 0);
  for (int s = 0; s < nsteps; ++s) {
    if (s + 1 < nsteps) {
      issue((s + 1) & 1, (s + 1) << 5);
      wait_async_le4();         // in-order: stage s has landed
    } else {
      wait_async_0();
    }
    __syncthreads();            // stage-s panels visible block-wide

    const _Float16* Ap = Ab[s & 1];
    const _Float16* Bp = Bb[s & 1];
    half16 af[2], bf[4];
#pragma unroll
    for (int i = 0; i < 2; ++i) {
      const _Float16* p = Ap + (wm * 32 + i * 16 + ar) * 32 + 8 * h;
      af[i] = cat16(*(const half8*)p, *(const half8*)(p + 16));
    }
#pragma unroll
    for (int j = 0; j < 4; ++j) {
      const _Float16* p = Bp + (wn * 64 + j * 16 + ar) * 32 + 8 * h;
      bf[j] = cat16(*(const half8*)p, *(const half8*)(p + 16));
    }
#pragma unroll
    for (int i = 0; i < 2; ++i)
#pragma unroll
      for (int j = 0; j < 4; ++j)
        acc[i][j] = wmma_f32_f16(af[i], bf[j], acc[i][j]);

    __syncthreads();            // everyone done reading before refill
  }

  // epilogue: bias (+ReLU), f32 and/or f16 store
#pragma unroll
  for (int j = 0; j < 4; ++j) {
    int col = bn0 + wn * 64 + j * 16 + ar;
    float bb = bias[col];
#pragma unroll
    for (int i = 0; i < 2; ++i) {
#pragma unroll
      for (int v = 0; v < 8; ++v) {
        float val = acc[i][j][v] + bb;
        if (relu) val = fmaxf(val, 0.0f);
        int row = tm0 + wm * 32 + i * 16 + v + 8 * h;
        size_t idx = (size_t)row * N + col;
        if (Cf) Cf[idx] = val;
        if (Ch) Ch[idx] = (_Float16)val;
      }
    }
  }
}

// ---------------------------------------------------------------------------
// Flash attention (unscaled dot product, per tf.keras reference).
// Block = 8 waves = 128 consecutive queries of one (batch, head); each wave
// owns a 16-query tile. Keys iterate in blocks of 32; the block stages each
// K tile [32 keys][64 depth] and V tile [64 depth][32 keys] into
// double-buffered LDS via async DMA (2 async b128 per thread per stage),
// shared by all 8 waves. qh/kh: f16 [B,S,DM] (head-sliced),
// vt: f16 [B,NH,DEPTH,S]. ctxh out: f16 [B,S,NH,DEPTH].
// ---------------------------------------------------------------------------
__global__ __launch_bounds__(256)
void flash_attn_kernel(const _Float16* __restrict__ qh,
                       const _Float16* __restrict__ kh,
                       const _Float16* __restrict__ vt,
                       _Float16*       __restrict__ ctxh) {
  __shared__ __align__(16) _Float16 Kb[2][32 * 64];    // [key][depth]
  __shared__ __align__(16) _Float16 Vb[2][64 * 32];    // [depth][key]
  __shared__ __align__(16) _Float16 pbuf[8][16 * 32];  // per-wave P tile

  int tid  = threadIdx.x;
  int w    = tid >> 5;
  int lane = tid & 31;
  int qblk = blockIdx.x % (SEQ / 128);
  int hd   = (blockIdx.x / (SEQ / 128)) % NH;
  int b    = blockIdx.x / ((SEQ / 128) * NH);
  int q0   = qblk * 128 + w * 16;
  int n = lane & 15, lh = lane >> 4;

  // async fill mapping: K tile rows of 128B (8 segs), V tile rows of 64B (4)
  int krow = tid >> 3, kseg = tid & 7;
  int vrow = tid >> 2, vseg = tid & 3;
  const _Float16* ksrc = kh + ((size_t)(b * SEQ)) * DM + hd * DEPTH;
  const _Float16* vsrc =
      vt + ((size_t)((b * NH + hd) * DEPTH) + vrow) * SEQ;

  auto issue = [&](int buf, int j) {
    const _Float16* gk = ksrc + (size_t)(j + krow) * DM + kseg * 8;
    const _Float16* gv = vsrc + j + vseg * 8;
    uint32_t lk = (uint32_t)(size_t)&Kb[buf][krow * 64 + kseg * 8];
    uint32_t lv = (uint32_t)(size_t)&Vb[buf][vrow * 32 + vseg * 8];
    async_b128(gk, lk);
    async_b128(gv, lv);
  };

  const _Float16* qbase = qh + ((size_t)(b * SEQ + q0)) * DM + hd * DEPTH;
  half16 qf0 = load_frag_a(qbase, DM, 0,  lane);   // depth 0..31
  half16 qf1 = load_frag_a(qbase, DM, 32, lane);   // depth 32..63

  v8f c0 = {}, c1 = {}, c2 = {}, c3 = {};          // ctx accum, 16 x 64
  float m[8], l[8];
#pragma unroll
  for (int v = 0; v < 8; ++v) { m[v] = -3.0e38f; l[v] = 0.0f; }

  constexpr int NSTEP = SEQ / 32;                  // 64
  issue(0, 0);
  for (int s = 0; s < NSTEP; ++s) {
    if (s + 1 < NSTEP) {
      issue((s + 1) & 1, (s + 1) * 32);
      wait_async_le2();          // in-order: stage s has landed
    } else {
      wait_async_0();
    }
    __syncthreads();             // stage-s K/V tiles visible block-wide

    const _Float16* Kp = Kb[s & 1];
    const _Float16* Vp = Vb[s & 1];

    // ---- scores: 16 queries x 32 keys (depth is WMMA K-dim) ----
    v8f s0 = {}, s1 = {};
    {
      const _Float16* p = Kp + n * 64 + 8 * lh;           // key tile 0
      s0 = wmma_f32_f16(qf0, cat16(*(const half8*)p,
                                   *(const half8*)(p + 16)), s0);
      s0 = wmma_f32_f16(qf1, cat16(*(const half8*)(p + 32),
                                   *(const half8*)(p + 48)), s0);
      const _Float16* q = Kp + (16 + n) * 64 + 8 * lh;    // key tile 1
      s1 = wmma_f32_f16(qf0, cat16(*(const half8*)q,
                                   *(const half8*)(q + 16)), s1);
      s1 = wmma_f32_f16(qf1, cat16(*(const half8*)(q + 32),
                                   *(const half8*)(q + 48)), s1);
    }

    // ---- online softmax (row = v + 8*lh, spread over 16-lane half) ----
#pragma unroll
    for (int v = 0; v < 8; ++v) {
      float mt = fmaxf(s0[v], s1[v]);
      for (int off = 1; off < 16; off <<= 1)
        mt = fmaxf(mt, __shfl_xor(mt, off, 16));
      float mn    = fmaxf(m[v], mt);
      float scale = __expf(m[v] - mn);
      float p0 = __expf(s0[v] - mn);
      float p1 = __expf(s1[v] - mn);
      float rs = p0 + p1;
      for (int off = 1; off < 16; off <<= 1)
        rs += __shfl_xor(rs, off, 16);
      l[v] = l[v] * scale + rs;
      m[v] = mn;
      c0[v] *= scale; c1[v] *= scale; c2[v] *= scale; c3[v] *= scale;
      int row = v + 8 * lh;
      pbuf[w][row * 32 + n]      = (_Float16)p0;
      pbuf[w][row * 32 + 16 + n] = (_Float16)p1;
    }

    // C-layout -> A-fragment layout via wave-private LDS (per-wave DS ops
    // are in-order; compiler inserts the s_wait_dscnt before use).
    half16 pf;
    {
      const _Float16* pp = &pbuf[w][(lane & 15) * 32 + 8 * lh];
      pf = cat16(*(const half8*)pp, *(const half8*)(pp + 16));
    }

    // ---- ctx += P(16x32) * V(32x64) from the shared V tile ----
#pragma unroll
    for (int t = 0; t < 4; ++t) {
      const _Float16* p = Vp + (16 * t + n) * 32 + 8 * lh;
      half16 vf = cat16(*(const half8*)p, *(const half8*)(p + 16));
      if      (t == 0) c0 = wmma_f32_f16(pf, vf, c0);
      else if (t == 1) c1 = wmma_f32_f16(pf, vf, c1);
      else if (t == 2) c2 = wmma_f32_f16(pf, vf, c2);
      else             c3 = wmma_f32_f16(pf, vf, c3);
    }

    __syncthreads();             // everyone done reading before refill
  }

#pragma unroll
  for (int v = 0; v < 8; ++v) {
    float inv = 1.0f / l[v];
    int row = q0 + v + 8 * lh;
    size_t base = ((size_t)(b * SEQ + row)) * DM + hd * DEPTH + n;
    ctxh[base + 0]  = (_Float16)(c0[v] * inv);
    ctxh[base + 16] = (_Float16)(c1[v] * inv);
    ctxh[base + 32] = (_Float16)(c2[v] * inv);
    ctxh[base + 48] = (_Float16)(c3[v] * inv);
  }
}

// ---------------------------------------------------------------------------
// y = LayerNorm(a + b) * g + be ; one row (512 cols) per block.
// ---------------------------------------------------------------------------
__global__ __launch_bounds__(256)
void add_ln_kernel(const float* __restrict__ a, const float* __restrict__ bsrc,
                   const float* __restrict__ g, const float* __restrict__ be,
                   float* __restrict__ of, _Float16* __restrict__ oh) {
  __shared__ float vals[DM];
  __shared__ float red[256];
  int row = blockIdx.x, tid = threadIdx.x;
  size_t base = (size_t)row * DM;

  float s = 0.0f;
  for (int i = tid; i < DM; i += 256) {
    float v = a[base + i] + bsrc[base + i];
    vals[i] = v;
    s += v;
  }
  red[tid] = s; __syncthreads();
  for (int st = 128; st > 0; st >>= 1) {
    if (tid < st) red[tid] += red[tid + st];
    __syncthreads();
  }
  float mu = red[0] * (1.0f / DM);
  __syncthreads();

  float s2 = 0.0f;
  for (int i = tid; i < DM; i += 256) {
    float d = vals[i] - mu;
    s2 += d * d;
  }
  red[tid] = s2; __syncthreads();
  for (int st = 128; st > 0; st >>= 1) {
    if (tid < st) red[tid] += red[tid + st];
    __syncthreads();
  }
  float var  = red[0] * (1.0f / DM);
  float rstd = rsqrtf(var + 1e-6f);

  for (int i = tid; i < DM; i += 256) {
    float y = (vals[i] - mu) * rstd * g[i] + be[i];
    of[base + i] = y;
    if (oh) oh[base + i] = (_Float16)y;
  }
}

// ---------------------------------------------------------------------------
// Host-side orchestration
// ---------------------------------------------------------------------------
extern "C" void kernel_launch(void* const* d_in, const int* in_sizes, int n_in,
                              void* d_out, int out_size, void* d_ws,
                              size_t ws_size, hipStream_t stream) {
  (void)in_sizes; (void)n_in; (void)out_size; (void)ws_size;

  const float* x   = (const float*)d_in[0];
  const float* Wq  = (const float*)d_in[1];
  const float* bq  = (const float*)d_in[2];
  const float* Wk  = (const float*)d_in[3];
  const float* bk  = (const float*)d_in[4];
  const float* Wv  = (const float*)d_in[5];
  const float* bv  = (const float*)d_in[6];
  const float* Wo  = (const float*)d_in[7];
  const float* bo  = (const float*)d_in[8];
  const float* W1  = (const float*)d_in[9];
  const float* b1  = (const float*)d_in[10];
  const float* W2  = (const float*)d_in[11];
  const float* b2  = (const float*)d_in[12];
  const float* g1  = (const float*)d_in[13];
  const float* be1 = (const float*)d_in[14];
  const float* g2  = (const float*)d_in[15];
  const float* be2 = (const float*)d_in[16];
  float* out = (float*)d_out;

  char* ws = (char*)d_ws;
  // byte offsets (with reuse; peak ~94 MB)
  const size_t off_xh   = 0;                        // 8 MB  f16 [8192][512]
  const size_t off_wqT  = 8388608;                  // 512KB f16 [512][512]
  const size_t off_wkT  = off_wqT + 524288;
  const size_t off_wvT  = off_wkT + 524288;
  const size_t off_woT  = off_wvT + 524288;
  const size_t off_w1T  = off_woT + 524288;         // 2 MB  f16 [2048][512]
  const size_t off_w2T  = off_w1T + 2097152;        // 2 MB  f16 [512][2048]
  const size_t off_qh   = off_w2T + 2097152;        // 8 MB
  const size_t off_kh   = off_qh + 8388608;         // 8 MB
  const size_t off_vh   = off_kh + 8388608;         // 8 MB
  const size_t off_vt   = off_vh + 8388608;         // 8 MB
  const size_t off_ctxh = off_vt + 8388608;         // 8 MB
  const size_t off_attn = off_ctxh + 8388608;       // 16 MB f32
  const size_t off_x1   = off_attn + 16777216;      // 16 MB f32
  const size_t off_x1h  = off_x1 + 16777216;        // 8 MB
  const size_t off_ffh  = off_qh;                   // 32 MB (reuses q/k/v/vt)
  const size_t off_ff2  = off_attn;                 // 16 MB (reuses attn)

  _Float16* xh   = (_Float16*)(ws + off_xh);
  _Float16* wqT  = (_Float16*)(ws + off_wqT);
  _Float16* wkT  = (_Float16*)(ws + off_wkT);
  _Float16* wvT  = (_Float16*)(ws + off_wvT);
  _Float16* woT  = (_Float16*)(ws + off_woT);
  _Float16* w1T  = (_Float16*)(ws + off_w1T);
  _Float16* w2T  = (_Float16*)(ws + off_w2T);
  _Float16* qh   = (_Float16*)(ws + off_qh);
  _Float16* kh   = (_Float16*)(ws + off_kh);
  _Float16* vh   = (_Float16*)(ws + off_vh);
  _Float16* vt   = (_Float16*)(ws + off_vt);
  _Float16* ctxh = (_Float16*)(ws + off_ctxh);
  float*    attn = (float*)(ws + off_attn);
  float*    x1   = (float*)(ws + off_x1);
  _Float16* x1h  = (_Float16*)(ws + off_x1h);
  _Float16* ffh  = (_Float16*)(ws + off_ffh);
  float*    ff2  = (float*)(ws + off_ff2);

  const int T = 256;
  const int nElems = ROWS * DM;

  // 1. convert activations + weights to f16 (weights transposed for B-frags)
  cvt_f16_kernel<<<nElems / T, T, 0, stream>>>(x, xh, nElems);
  transpose_cvt_kernel<<<(DM * DM) / T,  T, 0, stream>>>(Wq, wqT, DM, DM);
  transpose_cvt_kernel<<<(DM * DM) / T,  T, 0, stream>>>(Wk, wkT, DM, DM);
  transpose_cvt_kernel<<<(DM * DM) / T,  T, 0, stream>>>(Wv, wvT, DM, DM);
  transpose_cvt_kernel<<<(DM * DM) / T,  T, 0, stream>>>(Wo, woT, DM, DM);
  transpose_cvt_kernel<<<(DM * DFF) / T, T, 0, stream>>>(W1, w1T, DM, DFF);
  transpose_cvt_kernel<<<(DFF * DM) / T, T, 0, stream>>>(W2, w2T, DFF, DM);

  // 2. Q/K/V projections (f16 out); grid = (M/128)*(N/128) blocks
  int blk_dm = (ROWS / 128) * (DM / 128);          // 256
  gemm_f16_kernel<<<blk_dm, T, 0, stream>>>(xh, wqT, bq, nullptr, qh,
                                            ROWS, DM, DM, 0);
  gemm_f16_kernel<<<blk_dm, T, 0, stream>>>(xh, wkT, bk, nullptr, kh,
                                            ROWS, DM, DM, 0);
  gemm_f16_kernel<<<blk_dm, T, 0, stream>>>(xh, wvT, bv, nullptr, vh,
                                            ROWS, DM, DM, 0);

  // 3. V -> [B,NH,DEPTH,S] for contiguous B-fragments in P*V
  transpose_v_kernel<<<nElems / T, T, 0, stream>>>(vh, vt);

  // 4. flash attention: B*NH*(S/128) = 512 blocks of 8 waves
  flash_attn_kernel<<<512, T, 0, stream>>>(qh, kh, vt, ctxh);

  // 5. output projection (f32 out)
  gemm_f16_kernel<<<blk_dm, T, 0, stream>>>(ctxh, woT, bo, attn, nullptr,
                                            ROWS, DM, DM, 0);

  // 6. x1 = LN(x + attn) (f32 + f16 out)
  add_ln_kernel<<<ROWS, T, 0, stream>>>(x, attn, g1, be1, x1, x1h);

  // 7. ffh = relu(x1 @ W1 + b1) (f16 out)
  int blk_ff = (ROWS / 128) * (DFF / 128);         // 1024
  gemm_f16_kernel<<<blk_ff, T, 0, stream>>>(x1h, w1T, b1, nullptr, ffh,
                                            ROWS, DFF, DM, 1);

  // 8. ff2 = ffh @ W2 + b2 (f32 out)
  gemm_f16_kernel<<<blk_dm, T, 0, stream>>>(ffh, w2T, b2, ff2, nullptr,
                                            ROWS, DM, DFF, 0);

  // 9. out = LN(x1 + ff2)
  add_ln_kernel<<<ROWS, T, 0, stream>>>(x1, ff2, g2, be2, out, nullptr);
}
